// MeshGraphNet_29368986370151
// MI455X (gfx1250) — matrix operand order repaired and checked by reading, not compiled, not run on hardware
//
#include <hip/hip_runtime.h>
#include <hip/hip_bf16.h>
#include <stdint.h>

// ---------------------------------------------------------------------------
// MeshGraphNet forward for MI455X (gfx1250, wave32, WMMA).
//
// All GEMMs on v_wmma_f32_16x16x32_f16 (f16 A/B, f32 accumulate).
// Weights are pre-converted ONCE per launch (k_prep) into transposed, padded
// f16 blobs in workspace; compute blocks DMA the blob into LDS with the
// Tensor Data Mover (tensor_load_to_lds + s_wait_tensorcnt) when available,
// overlapping the weight copy with the activation gather. Fallback: b128
// copy loop.
// Per block: 256 threads = 8 wave32, TM=64 rows of the batch dimension.
// ---------------------------------------------------------------------------

typedef _Float16 h16;
typedef __attribute__((ext_vector_type(16))) _Float16 v16h;
typedef __attribute__((ext_vector_type(8)))  float    v8f;

#define THREADS 256
#define TM      64
#define SLOPE   0.01f
#define LN_EPS  1e-5f

// Weight blob layout per MLP "slot" (f16 elements):
//   W1t [64][KINp] | W2t [64][64] | W3t [NOUTp][64]
// Slot stride (max size, proc_e): 64*96 + 64*64 + 32*64 = 12288 elements.
#define WSLOT 12288

// ---------------------------------------------------------------------------
// Tensor Data Mover path (weight blob -> LDS), with portable fallback.
// ---------------------------------------------------------------------------
#if defined(__gfx1250__) && __has_builtin(__builtin_amdgcn_tensor_load_to_lds) && \
    __has_builtin(__builtin_amdgcn_s_wait_tensorcnt)
#define HAVE_TDM 1
#endif

#if __has_include(<hip/amd_detail/amd_gfx1250_TDM.h>)
#define TDM_ARGS6 1   // therock/clang-23 toolchain: 6-arg builtin
#endif

#ifdef HAVE_TDM
typedef unsigned int u32x4 __attribute__((ext_vector_type(4)));
typedef int          i32x4 __attribute__((ext_vector_type(4)));
typedef int          i32x8 __attribute__((ext_vector_type(8)));

// 1-D copy of `nelem` f16 from global `gsrc` into LDS byte offset `lds_off`.
// D# per CDNA5 ISA: group0 {count=1, lds_addr,
// global_addr, type=2}; group1 {data_size=2B, tensor_dim0=nelem,
// tensor_dim1=1, tile_dim0=nelem, tile_dim1=1}; groups 2/3 zero (<=2D).
static __device__ __forceinline__ void tdm_load_weights(const h16* gsrc,
                                                        unsigned lds_off,
                                                        int nelem) {
  const unsigned long long ga = (unsigned long long)(uintptr_t)gsrc;
  const unsigned ne = (unsigned)nelem;
  u32x4 g0 = {1u,                                   // count=1 (valid user D#)
              lds_off,                               // lds_addr (bytes)
              (unsigned)(ga & 0xFFFFFFFFu),          // global_addr[31:0]
              (unsigned)((ga >> 32) & 0x1FFFFFFu) | (2u << 30)};  // +type=2
  i32x8 g1 = {(int)(1u << 16),                       // data_size=1 (2 bytes)
              (int)((ne & 0xFFFFu) << 16),           // tensor_dim0[15:0]
              (int)((ne >> 16) | (1u << 16)),        // t_dim0[31:16]|t_dim1=1
              (int)(ne << 16),                       // t_dim1 hi=0 | tile_dim0
              1,                                     // tile_dim1=1, tile_dim2=0
              (int)ne,                               // tensor_dim0_stride lo
              0, 0};
  i32x4 z4 = {0, 0, 0, 0};
#ifdef TDM_ARGS6
  i32x8 z8 = {0, 0, 0, 0, 0, 0, 0, 0};
  __builtin_amdgcn_tensor_load_to_lds(g0, g1, z4, z4, z8, 0);
#else
  __builtin_amdgcn_tensor_load_to_lds(g0, g1, z4, z4, 0);
#endif
}
#endif  // HAVE_TDM

// Stage the per-MLP weight blob (nelem f16, 16B-multiple) into LDS at offset 0
// of the dynamic shared segment. TDM path: wave 0 issues the DMA and waits
// TENSORcnt; callers must __syncthreads() before using the data.
static __device__ __forceinline__ void stage_weights(h16* Wlds, const h16* Wt,
                                                     int nelem, int tid) {
#ifdef HAVE_TDM
  if (tid == 0) {
#if __has_builtin(__builtin_amdgcn_groupstaticsize)
    const unsigned base = __builtin_amdgcn_groupstaticsize();
#else
    const unsigned base = 0u;
#endif
    tdm_load_weights(Wt, base, nelem);
    __builtin_amdgcn_s_wait_tensorcnt(0);
  }
#else
  for (int i = tid; i < nelem / 8; i += THREADS)
    ((uint4*)Wlds)[i] = ((const uint4*)Wt)[i];
#endif
}

// ---------------------------------------------------------------------------
// WMMA fragments (ISA §7.12.2 wave32 layouts).
// ---------------------------------------------------------------------------
static __device__ __forceinline__ v8f wmma_f16f32(v16h a, v16h b, v8f c) {
  return __builtin_amdgcn_wmma_f32_16x16x32_f16(false, a, false, b, (short)0, c,
                                                false, false);
}

// A: 16x32 f16 tile from row-major LDS [rows][ldk].
// lanes 0-15 -> M=lane, K 0-15; lanes 16-31 -> M=lane-16,
// K 16-31.
static __device__ __forceinline__ v16h load_a_frag(const h16* __restrict__ A,
                                                   int ldk, int row0, int k0,
                                                   int lane) {
  const int hl = lane >> 4;
  const int r  = lane & 15;
  const h16* row = A + (row0 + r) * ldk;
  v16h a;
#pragma unroll
  for (int j = 0; j < 8; ++j) {
    const int kb = k0 + (hl << 4) + (j << 1);
    a[2 * j]     = row[kb];
    a[2 * j + 1] = row[kb + 1];
  }
  return a;
}

// B: 32x16 (KxN) tile; weights staged TRANSPOSED as Wt[N][ldk] so each lane
// (one N column) reads contiguous K pairs (merges into ds_load_b128).
static __device__ __forceinline__ v16h load_b_frag(const h16* __restrict__ Wt,
                                                   int ldk, int k0, int col0,
                                                   int lane) {
  const int c  = lane & 15;
  const int kg = (lane >> 4) << 4;
  const h16* col = Wt + (col0 + c) * ldk + k0 + kg;
  v16h b;
#pragma unroll
  for (int j = 0; j < 8; ++j) {
    b[2 * j]     = col[2 * j];
    b[2 * j + 1] = col[2 * j + 1];
  }
  return b;
}

// TM x NN = (TM x KK) @ (KK x NN) + bias, optional leaky-relu.
// Tiles enumerated column-major so each wave's M-tile is invariant:
// A fragments loaded once per wave, reused across its N-tiles.
template <int KK, int NN, bool LEAKY, bool OUT_F32>
static __device__ __forceinline__ void gemm_tile(const h16* __restrict__ A,
                                                 const h16* __restrict__ Wt,
                                                 const float* __restrict__ bias,
                                                 h16* __restrict__ outH,
                                                 float* __restrict__ outF,
                                                 int tid) {
  const int lane = tid & 31;
  const int wave = tid >> 5;
  const int hl   = lane >> 4;
  const int nn   = lane & 15;
  constexpr int MT = TM / 16;        // 4
  constexpr int NT = NN / 16;
  constexpr int NW = THREADS / 32;   // 8
  const int row0 = (wave & (MT - 1)) * 16;

  v16h afr[KK / 32];
#pragma unroll
  for (int k0 = 0; k0 < KK; k0 += 32)
    afr[k0 / 32] = load_a_frag(A, KK, row0, k0, lane);

  for (int nt = wave / MT; nt < NT; nt += NW / MT) {
    const int col0 = nt * 16;
    v8f acc = {};
#pragma unroll
    for (int k0 = 0; k0 < KK; k0 += 32)
      acc = wmma_f16f32(afr[k0 / 32], load_b_frag(Wt, KK, k0, col0, lane), acc);
#pragma unroll
    for (int v = 0; v < 8; ++v) {
      const int m = row0 + hl * 8 + v;
      const int c = col0 + nn;
      float val = acc[v] + bias[c];
      if (LEAKY) val = fmaxf(val, SLOPE * val);   // leaky-relu, slope 0.01
      if (OUT_F32) outF[m * NN + c] = val;
      else         outH[m * NN + c] = (h16)val;
    }
  }
}

// ---------------------------------------------------------------------------
// k_prep: convert one MLP's f32 weights into the transposed/padded f16 blob.
//   W1t[64][KINp] (K padded, zero-filled) | W2t[64][64] | W3t[NOUTp][64].
// ---------------------------------------------------------------------------
__global__ __launch_bounds__(THREADS) void k_prep(
    const float* __restrict__ Wi, const float* __restrict__ Wh,
    const float* __restrict__ Wo, h16* __restrict__ dst, int KINr, int KINp,
    int NOUTr, int NOUTp) {
  const int total = 64 * KINp + 4096 + 64 * NOUTp;
  const int i = blockIdx.x * THREADS + threadIdx.x;
  if (i >= total) return;
  const int o2 = 64 * KINp, o3 = o2 + 4096;
  if (i < o2) {
    const int nc = i / KINp, k = i - nc * KINp;
    dst[i] = (k < KINr) ? (h16)Wi[k * 64 + nc] : (h16)0.f;
  } else if (i < o3) {
    const int j = i - o2, nc = j >> 6, k = j & 63;
    dst[i] = (h16)Wh[k * 64 + nc];
  } else {
    const int j = i - o3, nc = j >> 6, k = j & 63;
    dst[i] = (nc < NOUTr) ? (h16)Wo[k * NOUTr + nc] : (h16)0.f;
  }
}

// ---------------------------------------------------------------------------
// Encoder: feats[N x INF] -> MLP(INF->64->64->32) -> LayerNorm -> out[N x 32]
// K padded to 32 (INF = 8 or 4). Weight blob: 64*32 + 4096 + 64*32 = 8192.
// ---------------------------------------------------------------------------
template <int INF>
__global__ __launch_bounds__(THREADS) void k_encode(
    const float* __restrict__ feats, const h16* __restrict__ Wt,
    const float* __restrict__ bi, const float* __restrict__ bh,
    const float* __restrict__ bo, const float* __restrict__ g,
    const float* __restrict__ bln, float* __restrict__ out, int n) {
  constexpr int NWH = 64 * 32 + 4096 + 64 * 32;  // 8192
  extern __shared__ char smraw[];
  char* sp = smraw;
  h16* Wlds = (h16*)sp;   sp += NWH * sizeof(h16);       // must be at offset 0
  h16* X    = (h16*)sp;   sp += TM * 32 * sizeof(h16);
  h16* H1   = (h16*)sp;   sp += TM * 64 * sizeof(h16);
  h16* H2   = (h16*)sp;   sp += TM * 64 * sizeof(h16);
  float* F  = (float*)sp; sp += TM * 32 * sizeof(float);
  float* biS = (float*)sp; sp += 64 * sizeof(float);
  float* bhS = (float*)sp; sp += 64 * sizeof(float);
  float* boS = (float*)sp; sp += 32 * sizeof(float);
  float* gS  = (float*)sp; sp += 32 * sizeof(float);
  float* bS  = (float*)sp; sp += 32 * sizeof(float);
  const h16* W1t = Wlds;              // [64][32]
  const h16* W2t = Wlds + 2048;       // [64][64]
  const h16* W3t = Wlds + 6144;       // [32][64]

  const int tid = threadIdx.x;
  const int r0  = blockIdx.x * TM;

  stage_weights(Wlds, Wt, NWH, tid);  // TDM DMA overlaps the gather below

  for (int i = tid; i < TM * 32; i += THREADS) {
    const int r = i >> 5, c = i & 31;
    const int row = r0 + r;
    X[i] = (h16)((row < n && c < INF) ? feats[row * INF + c] : 0.f);
  }
  if (tid < 64) { biS[tid] = bi[tid]; bhS[tid] = bh[tid]; }
  if (tid < 32) { boS[tid] = bo[tid]; gS[tid] = g[tid]; bS[tid] = bln[tid]; }
  __syncthreads();

  gemm_tile<32, 64, true, false>(X, W1t, biS, H1, nullptr, tid);
  __syncthreads();
  gemm_tile<64, 64, true, false>(H1, W2t, bhS, H2, nullptr, tid);
  __syncthreads();
  gemm_tile<64, 32, false, true>(H2, W3t, boS, nullptr, F, tid);
  __syncthreads();

  // LayerNorm over 32 features, 4 lanes per row (wave32 xor-shuffles).
  const int row = tid >> 2, q = tid & 3;
  float vb[8], s = 0.f, ss = 0.f;
#pragma unroll
  for (int j = 0; j < 8; ++j) {
    const float v = F[row * 32 + q * 8 + j];
    vb[j] = v; s += v; ss += v * v;
  }
  s  += __shfl_xor(s, 1);  s  += __shfl_xor(s, 2);
  ss += __shfl_xor(ss, 1); ss += __shfl_xor(ss, 2);
  const float mean = s * 0.03125f;
  const float rstd = rsqrtf(ss * 0.03125f - mean * mean + LN_EPS);
  const int rowg = r0 + row;
  if (rowg < n) {
#pragma unroll
    for (int j = 0; j < 8; ++j) {
      const int c = q * 8 + j;
      out[rowg * 32 + c] = (vb[j] - mean) * rstd * gS[c] + bS[c];
    }
  }
}

// ---------------------------------------------------------------------------
// Edge update: cat(edge, node[src], node[dst]) [E x 96] -> MLP -> LN
// -> residual -> edgeBuf; f32 atomic scatter-add of new edge into pe[dst].
// Weight blob: 64*96 + 4096 + 64*32 = 12288.
// ---------------------------------------------------------------------------
__global__ __launch_bounds__(THREADS) void k_edge(
    float* __restrict__ edgeBuf, const float* __restrict__ node,
    const int* __restrict__ src, const int* __restrict__ dstI,
    float* __restrict__ pe, const h16* __restrict__ Wt,
    const float* __restrict__ bi, const float* __restrict__ bh,
    const float* __restrict__ bo, const float* __restrict__ g,
    const float* __restrict__ bln, int E) {
  constexpr int NWH = 64 * 96 + 4096 + 64 * 32;  // 12288
  extern __shared__ char smraw[];
  char* sp = smraw;
  h16* Wlds = (h16*)sp;   sp += NWH * sizeof(h16);       // offset 0
  h16* X    = (h16*)sp;   sp += TM * 96 * sizeof(h16);   // aliased by F below
  h16* H1   = (h16*)sp;   sp += TM * 64 * sizeof(h16);
  h16* H2   = (h16*)sp;   sp += TM * 64 * sizeof(h16);
  float* Eres = (float*)sp; sp += TM * 32 * sizeof(float);
  float* biS = (float*)sp; sp += 64 * sizeof(float);
  float* bhS = (float*)sp; sp += 64 * sizeof(float);
  float* boS = (float*)sp; sp += 32 * sizeof(float);
  float* gS  = (float*)sp; sp += 32 * sizeof(float);
  float* bS  = (float*)sp; sp += 32 * sizeof(float);
  int* sidx  = (int*)sp;  sp += TM * sizeof(int);
  int* didx  = (int*)sp;  sp += TM * sizeof(int);
  const h16* W1t = Wlds;               // [64][96]
  const h16* W2t = Wlds + 6144;        // [64][64]
  const h16* W3t = Wlds + 10240;       // [32][64]
  float* F = (float*)X;  // X dead after GEMM1; 12288B region holds 8192B F

  const int tid = threadIdx.x;
  const int e0  = blockIdx.x * TM;

  stage_weights(Wlds, Wt, NWH, tid);  // TDM DMA overlaps the gather below

  // Speculative prefetch of the next tile's edge rows (global_prefetch_b8).
  __builtin_prefetch(&edgeBuf[(size_t)(e0 + TM) * 32 + tid], 0, 1);

  for (int r = tid; r < TM; r += THREADS) {
    const int e = e0 + r;
    sidx[r] = (e < E) ? src[e] : 0;
    didx[r] = (e < E) ? dstI[e] : 0;
  }
  if (tid < 64) { biS[tid] = bi[tid]; bhS[tid] = bh[tid]; }
  if (tid < 32) { boS[tid] = bo[tid]; gS[tid] = g[tid]; bS[tid] = bln[tid]; }
  __syncthreads();  // sidx/didx visible for the gather

  for (int i = tid; i < TM * 96; i += THREADS) {
    const int r = i / 96, c = i - r * 96;
    const int e = e0 + r;
    float v = 0.f;
    if (e < E) {
      if (c < 32) {
        v = edgeBuf[(size_t)e * 32 + c];
        Eres[r * 32 + c] = v;
      } else if (c < 64) {
        v = node[(size_t)sidx[r] * 32 + (c - 32)];
      } else {
        v = node[(size_t)didx[r] * 32 + (c - 64)];
      }
    }
    X[i] = (h16)v;
  }
  __syncthreads();

  gemm_tile<96, 64, true, false>(X, W1t, biS, H1, nullptr, tid);
  __syncthreads();
  gemm_tile<64, 64, true, false>(H1, W2t, bhS, H2, nullptr, tid);
  __syncthreads();
  gemm_tile<64, 32, false, true>(H2, W3t, boS, nullptr, F, tid);
  __syncthreads();

  const int row = tid >> 2, q = tid & 3;
  float vb[8], s = 0.f, ss = 0.f;
#pragma unroll
  for (int j = 0; j < 8; ++j) {
    const float v = F[row * 32 + q * 8 + j];
    vb[j] = v; s += v; ss += v * v;
  }
  s  += __shfl_xor(s, 1);  s  += __shfl_xor(s, 2);
  ss += __shfl_xor(ss, 1); ss += __shfl_xor(ss, 2);
  const float mean = s * 0.03125f;
  const float rstd = rsqrtf(ss * 0.03125f - mean * mean + LN_EPS);
  const int e = e0 + row;
  if (e < E) {
    const int d = didx[row];
#pragma unroll
    for (int j = 0; j < 8; ++j) {
      const int c = q * 8 + j;
      const float val =
          (vb[j] - mean) * rstd * gS[c] + bS[c] + Eres[row * 32 + c];
      edgeBuf[(size_t)e * 32 + c] = val;
      unsafeAtomicAdd(&pe[(size_t)d * 32 + c], val);  // global_atomic_add_f32
    }
  }
}

// ---------------------------------------------------------------------------
// Node update: cat(node, pe_sum) [N x 64] -> MLP -> LN -> residual -> nodeBuf.
// Weight blob: 64*64 + 4096 + 64*32 = 10240.
// ---------------------------------------------------------------------------
__global__ __launch_bounds__(THREADS) void k_node(
    float* __restrict__ nodeBuf, const float* __restrict__ pe,
    const h16* __restrict__ Wt, const float* __restrict__ bi,
    const float* __restrict__ bh, const float* __restrict__ bo,
    const float* __restrict__ g, const float* __restrict__ bln, int n) {
  constexpr int NWH = 64 * 64 + 4096 + 64 * 32;  // 10240
  extern __shared__ char smraw[];
  char* sp = smraw;
  h16* Wlds = (h16*)sp;   sp += NWH * sizeof(h16);       // offset 0
  h16* X    = (h16*)sp;   sp += TM * 64 * sizeof(h16);   // aliased by F below
  h16* H1   = (h16*)sp;   sp += TM * 64 * sizeof(h16);
  h16* H2   = (h16*)sp;   sp += TM * 64 * sizeof(h16);
  float* Nres = (float*)sp; sp += TM * 32 * sizeof(float);
  float* biS = (float*)sp; sp += 64 * sizeof(float);
  float* bhS = (float*)sp; sp += 64 * sizeof(float);
  float* boS = (float*)sp; sp += 32 * sizeof(float);
  float* gS  = (float*)sp; sp += 32 * sizeof(float);
  float* bS  = (float*)sp; sp += 32 * sizeof(float);
  const h16* W1t = Wlds;              // [64][64]
  const h16* W2t = Wlds + 4096;       // [64][64]
  const h16* W3t = Wlds + 8192;       // [32][64]
  float* F = (float*)X;

  const int tid = threadIdx.x;
  const int r0  = blockIdx.x * TM;

  stage_weights(Wlds, Wt, NWH, tid);

  for (int i = tid; i < TM * 64; i += THREADS) {
    const int r = i >> 6, c = i & 63;
    const int row = r0 + r;
    float v = 0.f;
    if (row < n) {
      if (c < 32) {
        v = nodeBuf[(size_t)row * 32 + c];
        Nres[r * 32 + c] = v;
      } else {
        v = pe[(size_t)row * 32 + (c - 32)];
      }
    }
    X[i] = (h16)v;
  }
  if (tid < 64) { biS[tid] = bi[tid]; bhS[tid] = bh[tid]; }
  if (tid < 32) { boS[tid] = bo[tid]; gS[tid] = g[tid]; bS[tid] = bln[tid]; }
  __syncthreads();

  gemm_tile<64, 64, true, false>(X, W1t, biS, H1, nullptr, tid);
  __syncthreads();
  gemm_tile<64, 64, true, false>(H1, W2t, bhS, H2, nullptr, tid);
  __syncthreads();
  gemm_tile<64, 32, false, true>(H2, W3t, boS, nullptr, F, tid);
  __syncthreads();

  const int row = tid >> 2, q = tid & 3;
  float vb[8], s = 0.f, ss = 0.f;
#pragma unroll
  for (int j = 0; j < 8; ++j) {
    const float v = F[row * 32 + q * 8 + j];
    vb[j] = v; s += v; ss += v * v;
  }
  s  += __shfl_xor(s, 1);  s  += __shfl_xor(s, 2);
  ss += __shfl_xor(ss, 1); ss += __shfl_xor(ss, 2);
  const float mean = s * 0.03125f;
  const float rstd = rsqrtf(ss * 0.03125f - mean * mean + LN_EPS);
  const int rowg = r0 + row;
  if (rowg < n) {
#pragma unroll
    for (int j = 0; j < 8; ++j) {
      const int c = q * 8 + j;
      nodeBuf[(size_t)rowg * 32 + c] =
          (vb[j] - mean) * rstd * gS[c] + bS[c] + Nres[row * 32 + c];
    }
  }
}

// ---------------------------------------------------------------------------
// Decoder: node[N x 32] -> MLP(32->64->64->2) -> out[N x 2] (no LN).
// Last GEMM's N padded to 16. Weight blob: 64*32 + 4096 + 64*16 = 7168.
// ---------------------------------------------------------------------------
__global__ __launch_bounds__(THREADS) void k_out(
    const float* __restrict__ node, const h16* __restrict__ Wt,
    const float* __restrict__ bi, const float* __restrict__ bh,
    const float* __restrict__ bo, float* __restrict__ out, int n) {
  constexpr int NWH = 64 * 32 + 4096 + 64 * 16;  // 7168
  extern __shared__ char smraw[];
  char* sp = smraw;
  h16* Wlds = (h16*)sp;   sp += NWH * sizeof(h16);       // offset 0
  h16* X    = (h16*)sp;   sp += TM * 32 * sizeof(h16);
  h16* H1   = (h16*)sp;   sp += TM * 64 * sizeof(h16);
  h16* H2   = (h16*)sp;   sp += TM * 64 * sizeof(h16);
  float* F  = (float*)sp; sp += TM * 16 * sizeof(float);
  float* biS  = (float*)sp; sp += 64 * sizeof(float);
  float* bhS  = (float*)sp; sp += 64 * sizeof(float);
  float* bo16 = (float*)sp; sp += 16 * sizeof(float);
  const h16* W1t = Wlds;              // [64][32]
  const h16* W2t = Wlds + 2048;       // [64][64]
  const h16* W3t = Wlds + 6144;       // [16][64]

  const int tid = threadIdx.x;
  const int r0  = blockIdx.x * TM;

  stage_weights(Wlds, Wt, NWH, tid);

  for (int i = tid; i < TM * 32; i += THREADS) {
    const int r = i >> 5, c = i & 31;
    const int row = r0 + r;
    X[i] = (h16)((row < n) ? node[(size_t)row * 32 + c] : 0.f);
  }
  if (tid < 64) { biS[tid] = bi[tid]; bhS[tid] = bh[tid]; }
  if (tid < 16) { bo16[tid] = (tid < 2) ? bo[tid] : 0.f; }
  __syncthreads();

  gemm_tile<32, 64, true, false>(X, W1t, biS, H1, nullptr, tid);
  __syncthreads();
  gemm_tile<64, 64, true, false>(H1, W2t, bhS, H2, nullptr, tid);
  __syncthreads();
  gemm_tile<64, 16, false, true>(H2, W3t, bo16, nullptr, F, tid);
  __syncthreads();

  if (tid < TM * 2) {
    const int r = tid >> 1, c = tid & 1;
    const int row = r0 + r;
    if (row < n) out[(size_t)row * 2 + c] = F[r * 16 + c];
  }
}

__global__ __launch_bounds__(THREADS) void k_zero(float* __restrict__ p, int n) {
  const int i = blockIdx.x * THREADS + threadIdx.x;
  if (i < n) p[i] = 0.f;
}

// ---------------------------------------------------------------------------
// Host: param unflattening, weight prep, launch sequence.
// ---------------------------------------------------------------------------
static constexpr size_t SMEM_ENC =
    8192 * 2 + TM * 32 * 2 + TM * 64 * 2 + TM * 64 * 2 + TM * 32 * 4 +
    (64 + 64 + 32 + 32 + 32) * 4;
static constexpr size_t SMEM_EDGE =
    12288 * 2 + TM * 96 * 2 + TM * 64 * 2 + TM * 64 * 2 + TM * 32 * 4 +
    (64 + 64 + 32 + 32 + 32) * 4 + 2 * TM * 4;
static constexpr size_t SMEM_NODE =
    10240 * 2 + TM * 64 * 2 + TM * 64 * 2 + TM * 64 * 2 + TM * 32 * 4 +
    (64 + 64 + 32 + 32 + 32) * 4;
static constexpr size_t SMEM_OUT =
    7168 * 2 + TM * 32 * 2 + TM * 64 * 2 + TM * 64 * 2 + TM * 16 * 4 +
    (64 + 64 + 16) * 4;

extern "C" void kernel_launch(void* const* d_in, const int* in_sizes, int n_in,
                              void* d_out, int out_size, void* d_ws,
                              size_t ws_size, hipStream_t stream) {
  const float* nfeat = (const float*)d_in[0];
  const float* efeat = (const float*)d_in[1];
  const int*   src   = (const int*)d_in[2];
  const int*   dst   = (const int*)d_in[3];
  const int N = in_sizes[0] / 8;
  const int E = in_sizes[1] / 4;

  struct Mlp { const float *Wi, *bi, *Wh, *bh, *Wo, *bo, *g, *b; };
  auto get = [&](int i) { return (const float*)d_in[i]; };

  Mlp enc_n, enc_e, outp, proc_e[5], proc_n[5];
  if (in_sizes[4] == 8 * 64) {
    // Insertion-order flatten: params{enc_n, enc_e, proc_e[5], proc_n[5], out}
    // mlp leaves: Wi, bi, Wh0, bh0, Wo, bo [, g, b]
    auto mk = [&](int base, bool lnorm) {
      Mlp m;
      m.Wi = get(base + 0); m.bi = get(base + 1);
      m.Wh = get(base + 2); m.bh = get(base + 3);
      m.Wo = get(base + 4); m.bo = get(base + 5);
      m.g = lnorm ? get(base + 6) : nullptr;
      m.b = lnorm ? get(base + 7) : nullptr;
      return m;
    };
    enc_n = mk(4, true);
    enc_e = mk(12, true);
    for (int i = 0; i < 5; ++i) proc_e[i] = mk(20 + 8 * i, true);
    for (int i = 0; i < 5; ++i) proc_n[i] = mk(60 + 8 * i, true);
    outp = mk(100, false);
  } else {
    // jax sorted-key flatten: params{enc_e, enc_n, out, proc_e[5], proc_n[5]}
    // mlp(LN): Wh0, Wi, Wo, b, bh0, bi, bo, g ; mlp: Wh0, Wi, Wo, bh0, bi, bo
    auto mkln = [&](int base) {
      Mlp m;
      m.Wh = get(base + 0); m.Wi = get(base + 1); m.Wo = get(base + 2);
      m.b  = get(base + 3); m.bh = get(base + 4); m.bi = get(base + 5);
      m.bo = get(base + 6); m.g  = get(base + 7);
      return m;
    };
    auto mkpl = [&](int base) {
      Mlp m;
      m.Wh = get(base + 0); m.Wi = get(base + 1); m.Wo = get(base + 2);
      m.bh = get(base + 3); m.bi = get(base + 4); m.bo = get(base + 5);
      m.g = nullptr; m.b = nullptr;
      return m;
    };
    enc_e = mkln(4);
    enc_n = mkln(12);
    outp  = mkpl(20);
    for (int i = 0; i < 5; ++i) proc_e[i] = mkln(26 + 8 * i);
    for (int i = 0; i < 5; ++i) proc_n[i] = mkln(66 + 8 * i);
  }

  // Workspace: 13 f16 weight slots | node[N x 32] | pe[N x 32] | edge[E x 32].
  h16* wsl = (h16*)d_ws;
  float* nodeB = (float*)((char*)d_ws + (size_t)13 * WSLOT * sizeof(h16));
  float* peB   = nodeB + (size_t)N * 32;
  float* edgeB = peB + (size_t)N * 32;

  const dim3 blk(THREADS);
  const int gn = (N + TM - 1) / TM;
  const int ge = (E + TM - 1) / TM;
  const int gz = (N * 32 + THREADS - 1) / THREADS;

  // Weight prep: slot 0 enc_n, 1 enc_e, 2..6 proc_e, 7..11 proc_n, 12 out.
  auto prep = [&](const Mlp& m, int slot, int KINr, int KINp, int NOUTr,
                  int NOUTp) {
    const int total = 64 * KINp + 4096 + 64 * NOUTp;
    k_prep<<<(total + THREADS - 1) / THREADS, blk, 0, stream>>>(
        m.Wi, m.Wh, m.Wo, wsl + (size_t)slot * WSLOT, KINr, KINp, NOUTr,
        NOUTp);
  };
  prep(enc_n, 0, 8, 32, 32, 32);
  prep(enc_e, 1, 4, 32, 32, 32);
  for (int i = 0; i < 5; ++i) prep(proc_e[i], 2 + i, 96, 96, 32, 32);
  for (int i = 0; i < 5; ++i) prep(proc_n[i], 7 + i, 64, 64, 32, 32);
  prep(outp, 12, 32, 32, 2, 16);

  k_encode<8><<<gn, blk, SMEM_ENC, stream>>>(
      nfeat, wsl + 0 * WSLOT, enc_n.bi, enc_n.bh, enc_n.bo, enc_n.g, enc_n.b,
      nodeB, N);
  k_encode<4><<<ge, blk, SMEM_ENC, stream>>>(
      efeat, wsl + 1 * WSLOT, enc_e.bi, enc_e.bh, enc_e.bo, enc_e.g, enc_e.b,
      edgeB, E);

  for (int it = 0; it < 5; ++it) {
    k_zero<<<gz, blk, 0, stream>>>(peB, N * 32);
    k_edge<<<ge, blk, SMEM_EDGE, stream>>>(
        edgeB, nodeB, src, dst, peB, wsl + (size_t)(2 + it) * WSLOT,
        proc_e[it].bi, proc_e[it].bh, proc_e[it].bo, proc_e[it].g,
        proc_e[it].b, E);
    k_node<<<gn, blk, SMEM_NODE, stream>>>(
        nodeB, peB, wsl + (size_t)(7 + it) * WSLOT, proc_n[it].bi,
        proc_n[it].bh, proc_n[it].bo, proc_n[it].g, proc_n[it].b, N);
  }

  k_out<<<gn, blk, SMEM_OUT, stream>>>(
      nodeB, wsl + 12 * WSLOT, outp.bi, outp.bh, outp.bo, (float*)d_out, N);
}
